// SelfReferentialTitans_62637803044872
// MI455X (gfx1250) — matrix operand order repaired and checked by reading, not compiled, not run on hardware
//
#include <hip/hip_runtime.h>

// ---------------------------------------------------------------------------
// SelfReferentialTitans on MI455X (gfx1250).
// Math: the scan's weight update is w*a_t - 0.01*e_t with SCALAR a,e, so
//   w1_t = c_t*W1 + d_t*ones,  w2_t = c_t*W2 + d_t*ones
// and the whole model collapses to 7 GEMMs (16384x512x512) + elementwise ops.
// GEMMs use v_wmma_f32_16x16x32_bf16 with LDS double buffering; fragments are
// stored in LDS pre-swizzled to the ISA VGPR layout (2x b128 per lane).
// Global->LDS staging uses GLOBAL_LOAD_ASYNC_TO_LDS_B128 (ASYNCcnt-tracked).
// ---------------------------------------------------------------------------

typedef __bf16 bf16;
typedef __attribute__((ext_vector_type(16))) __bf16 v16bf;
typedef __attribute__((ext_vector_type(8)))  float  v8f;

#define D_DIM 512
#define L_DIM 2048
#define B_DIM 8
#define M_DIM (B_DIM * L_DIM)        // 16384 rows
#define NK_STEPS (D_DIM / 32)        // 16 k-steps of 32

#if defined(__has_builtin)
#  if __has_builtin(__builtin_amdgcn_global_load_async_to_lds_b128)
#    define USE_ASYNC_LDS 1
#  endif
#  if __has_builtin(__builtin_amdgcn_s_wait_asynccnt)
#    define HAVE_WAIT_ASYNC_BUILTIN 1
#  endif
#endif

#if defined(USE_ASYNC_LDS)
typedef int v4i __attribute__((vector_size(16)));
typedef __attribute__((address_space(1))) v4i Gv4i;   // global int4
typedef __attribute__((address_space(3))) v4i Lv4i;   // LDS int4
__device__ __forceinline__ void async_cp16(void* lds, const void* g) {
  // generic LDS pointer: low 32 bits are the LDS byte offset (ISA aperture map)
  __builtin_amdgcn_global_load_async_to_lds_b128(
      (Gv4i*)(uintptr_t)g,
      (Lv4i*)(unsigned)(uintptr_t)lds, 0, 0);
}
__device__ __forceinline__ void wait_async0() {
#  if defined(HAVE_WAIT_ASYNC_BUILTIN)
  __builtin_amdgcn_s_wait_asynccnt(0);
#  else
  asm volatile("s_wait_asynccnt 0x0" ::: "memory");
#  endif
}
#endif

__device__ __forceinline__ float gelu_exact(float x) {
  return 0.5f * x * (1.0f + erff(x * 0.70710678118654752f));
}
__device__ __forceinline__ float sigmoid_f(float x) {
  return 1.0f / (1.0f + expf(-x));
}

// ---------------------------------------------------------------------------
// Tiled bf16 WMMA GEMM: C(M x 512) = A(M x 512, bf16) * Bt^T, where Bt is the
// operand stored N-major (Bt[n][k], 512x512 bf16). Block tile 128x128, 8 waves
// (4 M x 2 N), each wave 32x64 via 2x4 wmma accumulators.
// EPI: 0 store f32 | 1 gelu->bf16 | 2 sigmoid(C+add) row-sum -> atomic per l |
//      3 h=gelu(c*C+d*s)->bf16 + rowsum->hs | 4 out = add + c*C + d*hs
// ---------------------------------------------------------------------------
template <int EPI>
__global__ void __launch_bounds__(256, 1)
gemm512(const bf16* __restrict__ A, const bf16* __restrict__ Bt,
        float* __restrict__ Cf, bf16* __restrict__ Cbf,
        const float* __restrict__ addF, const float* __restrict__ cvec,
        const float* __restrict__ dvec, const float* __restrict__ svec,
        float* __restrict__ redOut)
{
  // Fragment-order LDS: [buf][subtile(8)][lane(32)][2 x uint4]  (16KB each)
  __shared__ __align__(32) uint4 Alds[2][8][32][2];
  __shared__ __align__(32) uint4 Blds[2][8][32][2];

  const int tid  = threadIdx.x;
  const int lane = tid & 31;
  const int wave = tid >> 5;
  const int wm   = wave >> 1;                 // 0..3  (M direction)
  const int wn   = wave & 1;                  // 0..1  (N direction)
  const int m0   = blockIdx.y * 128;
  const int n0   = blockIdx.x * 128;

  const uint4* Ag = reinterpret_cast<const uint4*>(A);   // row stride 64 uint4
  const uint4* Bg = reinterpret_cast<const uint4*>(Bt);

  const v8f vz = {0.f,0.f,0.f,0.f,0.f,0.f,0.f,0.f};
  v8f acc[2][4];
#pragma unroll
  for (int i = 0; i < 2; ++i)
#pragma unroll
    for (int j = 0; j < 4; ++j) acc[i][j] = vz;

#if defined(USE_ASYNC_LDS)
  // ---- async DMA staging (GLOBAL_LOAD_ASYNC_TO_LDS_B128, ASYNCcnt) ----
  {
#pragma unroll
    for (int i = 0; i < 2; ++i) {
      int id = tid + 256 * i;
      int r = id >> 2, kk = id & 3;
      async_cp16(&Alds[0][r >> 4][(r & 15) + 16 * (kk & 1)][kk >> 1],
                 Ag + (size_t)(m0 + r) * 64 + kk);
      async_cp16(&Blds[0][r >> 4][(r & 15) + 16 * (kk & 1)][kk >> 1],
                 Bg + (size_t)(n0 + r) * 64 + kk);
    }
  }
  wait_async0();
  __syncthreads();

  for (int kt = 0; kt < NK_STEPS; ++kt) {
    const int buf = kt & 1;
    if (kt + 1 < NK_STEPS) {
#pragma unroll
      for (int i = 0; i < 2; ++i) {
        int id = tid + 256 * i;
        int r = id >> 2, kk = id & 3;
        async_cp16(&Alds[buf ^ 1][r >> 4][(r & 15) + 16 * (kk & 1)][kk >> 1],
                   Ag + (size_t)(m0 + r) * 64 + (kt + 1) * 4 + kk);
        async_cp16(&Blds[buf ^ 1][r >> 4][(r & 15) + 16 * (kk & 1)][kk >> 1],
                   Bg + (size_t)(n0 + r) * 64 + (kt + 1) * 4 + kk);
      }
    }

    v16bf af[2], bv[4];
#pragma unroll
    for (int i = 0; i < 2; ++i)
      af[i] = *reinterpret_cast<const v16bf*>(&Alds[buf][wm * 2 + i][lane][0]);
#pragma unroll
    for (int j = 0; j < 4; ++j)
      bv[j] = *reinterpret_cast<const v16bf*>(&Blds[buf][wn * 4 + j][lane][0]);

#pragma unroll
    for (int i = 0; i < 2; ++i)
#pragma unroll
      for (int j = 0; j < 4; ++j)
        acc[i][j] = __builtin_amdgcn_wmma_f32_16x16x32_bf16(
            false, af[i], false, bv[j], (short)0, acc[i][j], false, false);

    if (kt + 1 < NK_STEPS) wait_async0();
    __syncthreads();
  }
#else
  // ---- fallback: register-staged global->LDS ----
  uint4 ra[2], rb[2];
#pragma unroll
  for (int i = 0; i < 2; ++i) {
    int id = tid + 256 * i;
    int r = id >> 2, kk = id & 3;
    ra[i] = Ag[(size_t)(m0 + r) * 64 + kk];
    rb[i] = Bg[(size_t)(n0 + r) * 64 + kk];
  }

  for (int kt = 0; kt < NK_STEPS; ++kt) {
    const int buf = kt & 1;
#pragma unroll
    for (int i = 0; i < 2; ++i) {
      int id = tid + 256 * i;
      int r = id >> 2, kk = id & 3;
      Alds[buf][r >> 4][(r & 15) + 16 * (kk & 1)][kk >> 1] = ra[i];
      Blds[buf][r >> 4][(r & 15) + 16 * (kk & 1)][kk >> 1] = rb[i];
    }
    __syncthreads();

    if (kt + 1 < NK_STEPS) {
#pragma unroll
      for (int i = 0; i < 2; ++i) {
        int id = tid + 256 * i;
        int r = id >> 2, kk = id & 3;
        ra[i] = Ag[(size_t)(m0 + r) * 64 + (kt + 1) * 4 + kk];
        rb[i] = Bg[(size_t)(n0 + r) * 64 + (kt + 1) * 4 + kk];
      }
      if (kt + 2 < NK_STEPS) {
        int r = tid >> 2, kk = tid & 3;
        __builtin_prefetch(&Ag[(size_t)(m0 + r) * 64 + (kt + 2) * 4 + kk], 0, 1);
        __builtin_prefetch(&Bg[(size_t)(n0 + r) * 64 + (kt + 2) * 4 + kk], 0, 1);
      }
    }

    v16bf af[2], bv[4];
#pragma unroll
    for (int i = 0; i < 2; ++i)
      af[i] = *reinterpret_cast<const v16bf*>(&Alds[buf][wm * 2 + i][lane][0]);
#pragma unroll
    for (int j = 0; j < 4; ++j)
      bv[j] = *reinterpret_cast<const v16bf*>(&Blds[buf][wn * 4 + j][lane][0]);

#pragma unroll
    for (int i = 0; i < 2; ++i)
#pragma unroll
      for (int j = 0; j < 4; ++j)
        acc[i][j] = __builtin_amdgcn_wmma_f32_16x16x32_bf16(
            false, af[i], false, bv[j], (short)0, acc[i][j], false, false);
    __syncthreads();
  }
#endif

  // ---- epilogue (C/D layout: N = lane&15; M = v + 8*(lane>>4)) ----
  const int lh  = lane & 15;
  const int grp = lane >> 4;
#pragma unroll
  for (int i = 0; i < 2; ++i) {
#pragma unroll
    for (int v = 0; v < 8; ++v) {
      const int m = m0 + (wm * 2 + i) * 16 + grp * 8 + v;
      const int l = m & (L_DIM - 1);
      float rowacc = 0.f;
      float cl = 0.f, dl = 0.f, sv = 0.f;
      if (EPI == 3 || EPI == 4) { cl = cvec[l]; dl = dvec[l]; sv = svec[m]; }
#pragma unroll
      for (int j = 0; j < 4; ++j) {
        const int col = n0 + wn * 64 + j * 16 + lh;
        const size_t idx = (size_t)m * D_DIM + col;
        const float val = acc[i][j][v];
        if (EPI == 0) {
          Cf[idx] = val;
        } else if (EPI == 1) {
          Cbf[idx] = (bf16)gelu_exact(val);
        } else if (EPI == 2) {
          rowacc += sigmoid_f(val + addF[idx]);
        } else if (EPI == 3) {
          float h = gelu_exact(cl * val + dl * sv);
          Cbf[idx] = (bf16)h;
          rowacc += h;
        } else if (EPI == 4) {
          Cf[idx] = addF[idx] + cl * val + dl * sv;
        }
      }
      if (EPI == 2 || EPI == 3) {
        rowacc += __shfl_xor(rowacc, 1, 16);
        rowacc += __shfl_xor(rowacc, 2, 16);
        rowacc += __shfl_xor(rowacc, 4, 16);
        rowacc += __shfl_xor(rowacc, 8, 16);
        if (lh == 0) atomicAdd(&redOut[(EPI == 2) ? l : m], rowacc);
      }
    }
  }
}

// ---------------------------------------------------------------------------
// Causal depthwise conv1d (k=4, left pad 3) -> xc (f32 + bf16 copy)
// ---------------------------------------------------------------------------
__global__ void conv_kernel(const float* __restrict__ x, const float* __restrict__ w,
                            const float* __restrict__ b, float* __restrict__ xc,
                            bf16* __restrict__ xcbf)
{
  const size_t idx = (size_t)blockIdx.x * 256 + threadIdx.x;   // over M*D
  const int dch = (int)(idx & (D_DIM - 1));
  const size_t m = idx >> 9;
  const int l = (int)(m & (L_DIM - 1));
  float acc = b[dch];
#pragma unroll
  for (int j = 0; j < 4; ++j) {
    int lp = l - 3 + j;
    if (lp >= 0) acc += w[dch * 4 + j] * x[(m - l + lp) * D_DIM + dch];
  }
  xc[idx] = acc;
  xcbf[idx] = (bf16)acc;
}

// Convert a 512x512 f32 weight into bf16 B-operand, N-major (Bt[n][k]).
__global__ void prep_w(const float* __restrict__ src, bf16* __restrict__ dst, int transpose)
{
  const int idx = blockIdx.x * 256 + threadIdx.x;   // 512*512
  const int n = idx >> 9, k = idx & 511;
  float v = transpose ? src[k * D_DIM + n] : src[idx];
  dst[idx] = (bf16)v;
}

__global__ void zero_kernel(float* __restrict__ eta_sum, float* __restrict__ alpha_sum,
                            float* __restrict__ hs)
{
  const int i = blockIdx.x * 256 + threadIdx.x;
  if (i < L_DIM) { eta_sum[i] = 0.f; alpha_sum[i] = 0.f; }
  if (i < M_DIM) hs[i] = 0.f;
}

// l2-normalize each 512-row of q in place; emit bf16 copy and s[m]=rowsum(q).
__global__ void l2norm_kernel(float* __restrict__ q, bf16* __restrict__ qbf,
                              float* __restrict__ s)
{
  __shared__ float r0[256];
  __shared__ float r1[256];
  __shared__ float inv_s;
  const int m = blockIdx.x;
  const int t = threadIdx.x;
  const size_t base = (size_t)m * D_DIM;
  float v0 = q[base + t];
  float v1 = q[base + t + 256];
  r0[t] = v0 * v0 + v1 * v1;
  r1[t] = v0 + v1;
  __syncthreads();
  for (int off = 128; off > 0; off >>= 1) {
    if (t < off) { r0[t] += r0[t + off]; r1[t] += r1[t + off]; }
    __syncthreads();
  }
  if (t == 0) {
    float inv = 1.0f / fmaxf(sqrtf(r0[0]), 1e-12f);
    inv_s = inv;
    s[m] = r1[0] * inv;
  }
  __syncthreads();
  const float inv = inv_s;
  q[base + t]       = v0 * inv;  qbf[base + t]       = (bf16)(v0 * inv);
  q[base + t + 256] = v1 * inv;  qbf[base + t + 256] = (bf16)(v1 * inv);
}

// Scalar scan over L: c[t], d[t] are pre-update coefficients.
__global__ void scan_kernel(const float* __restrict__ eta_sum,
                            const float* __restrict__ alpha_sum,
                            float* __restrict__ cvec, float* __restrict__ dvec)
{
  if (blockIdx.x == 0 && threadIdx.x == 0) {
    const float invBD = 1.0f / (float)(B_DIM * D_DIM);
    float c = 1.0f, d = 0.0f;
    for (int t = 0; t < L_DIM; ++t) {
      cvec[t] = c; dvec[t] = d;
      float a = alpha_sum[t] * invBD;
      float e = eta_sum[t] * invBD;
      c = c * a;
      d = d * a - 0.01f * e;
    }
  }
}

// ---------------------------------------------------------------------------
extern "C" void kernel_launch(void* const* d_in, const int* in_sizes, int n_in,
                              void* d_out, int out_size, void* d_ws, size_t ws_size,
                              hipStream_t stream)
{
  (void)in_sizes; (void)n_in; (void)out_size; (void)ws_size;
  const float* x        = (const float*)d_in[0];
  const float* conv_w   = (const float*)d_in[1];
  const float* conv_b   = (const float*)d_in[2];
  const float* q_w      = (const float*)d_in[3];
  const float* mem_w1   = (const float*)d_in[4];
  const float* mem_w2   = (const float*)d_in[5];
  const float* eta_w1   = (const float*)d_in[6];
  const float* eta_w2   = (const float*)d_in[7];
  const float* alpha_w1 = (const float*)d_in[8];
  const float* alpha_w2 = (const float*)d_in[9];

  char* ws = (char*)d_ws;
  size_t off = 0;
  auto take = [&](size_t bytes) -> char* {
    char* p = ws + off;
    off += (bytes + 255) & ~(size_t)255;
    return p;
  };
  const size_t WB  = (size_t)D_DIM * D_DIM * sizeof(bf16);
  const size_t MF  = (size_t)M_DIM * D_DIM * sizeof(float);
  const size_t MBF = (size_t)M_DIM * D_DIM * sizeof(bf16);

  bf16* Wq   = (bf16*)take(WB);
  bf16* We1  = (bf16*)take(WB);
  bf16* We2  = (bf16*)take(WB);
  bf16* Wa1  = (bf16*)take(WB);
  bf16* Wa2  = (bf16*)take(WB);
  bf16* Wm1  = (bf16*)take(WB);
  bf16* Wm2  = (bf16*)take(WB);
  float* xc  = (float*)take(MF);
  bf16* xcbf = (bf16*)take(MBF);
  bf16* ge1  = (bf16*)take(MBF);
  bf16* ga1  = (bf16*)take(MBF);
  float* qf  = (float*)take(MF);
  bf16* qbf  = (bf16*)take(MBF);
  bf16* hbf  = (bf16*)take(MBF);
  float* svec      = (float*)take(M_DIM * sizeof(float));
  float* hs        = (float*)take(M_DIM * sizeof(float));
  float* eta_sum   = (float*)take(L_DIM * sizeof(float));
  float* alpha_sum = (float*)take(L_DIM * sizeof(float));
  float* cvec      = (float*)take(L_DIM * sizeof(float));
  float* dvec      = (float*)take(L_DIM * sizeof(float));

  const dim3 blk(256);
  const dim3 gw(1024);                    // 512*512/256
  const dim3 gg(D_DIM / 128, M_DIM / 128); // 4 x 128 GEMM blocks

  zero_kernel<<<dim3(M_DIM / 256), blk, 0, stream>>>(eta_sum, alpha_sum, hs);

  prep_w<<<gw, blk, 0, stream>>>(q_w,      Wq,  0);  // Bt[n][k] = q_w[n][k]
  prep_w<<<gw, blk, 0, stream>>>(eta_w1,   We1, 1);
  prep_w<<<gw, blk, 0, stream>>>(eta_w2,   We2, 1);
  prep_w<<<gw, blk, 0, stream>>>(alpha_w1, Wa1, 1);
  prep_w<<<gw, blk, 0, stream>>>(alpha_w2, Wa2, 1);
  prep_w<<<gw, blk, 0, stream>>>(mem_w1,   Wm1, 1);
  prep_w<<<gw, blk, 0, stream>>>(mem_w2,   Wm2, 1);

  conv_kernel<<<dim3((M_DIM * D_DIM) / 256), blk, 0, stream>>>(x, conv_w, conv_b, xc, xcbf);

  // eta/alpha first GEMM + fused exact-gelu -> bf16
  gemm512<1><<<gg, blk, 0, stream>>>(xcbf, We1, nullptr, ge1, nullptr, nullptr, nullptr, nullptr, nullptr);
  gemm512<1><<<gg, blk, 0, stream>>>(xcbf, Wa1, nullptr, ga1, nullptr, nullptr, nullptr, nullptr, nullptr);
  // second GEMM + fused sigmoid(xc + .) and per-l mean accumulation
  gemm512<2><<<gg, blk, 0, stream>>>(ge1, We2, nullptr, nullptr, xc, nullptr, nullptr, nullptr, eta_sum);
  gemm512<2><<<gg, blk, 0, stream>>>(ga1, Wa2, nullptr, nullptr, xc, nullptr, nullptr, nullptr, alpha_sum);
  // q projection (raw), then l2 normalize + rowsum
  gemm512<0><<<gg, blk, 0, stream>>>(xcbf, Wq, qf, nullptr, nullptr, nullptr, nullptr, nullptr, nullptr);
  l2norm_kernel<<<dim3(M_DIM), blk, 0, stream>>>(qf, qbf, svec);
  // scalar affine scan over L for (c_t, d_t)
  scan_kernel<<<dim3(1), dim3(32), 0, stream>>>(eta_sum, alpha_sum, cvec, dvec);
  // h = gelu(c*(q@W1) + d*s) -> bf16, rowsum -> hs
  gemm512<3><<<gg, blk, 0, stream>>>(qbf, Wm1, nullptr, hbf, nullptr, cvec, dvec, svec, hs);
  // out = q + c*(h@W2) + d*hs
  gemm512<4><<<gg, blk, 0, stream>>>(hbf, Wm2, (float*)d_out, nullptr, qf, cvec, dvec, hs, nullptr);
}
